// DeformableFeatureAggregationOE_83021717832259
// MI455X (gfx1250) — compile-verified, matrix operand order checked by reading, not run on hardware
//
#include <hip/hip_runtime.h>
#include <hip/hip_bf16.h>
#include <math.h>

// ---- problem constants (from reference) ----
#define NCAMS 6
#define NLVLS 4
#define NPTS  13
#define NGRP  8
#define BS    2
#define NA_Q  900
#define E_DIM 256
#define NSLOT (NCAMS * NLVLS * NPTS)   // 312
#define NWFC  (NSLOT * NGRP)           // 2496 (multiple of 64)
#define M_ROWS (BS * NA_Q)             // 1800
#define KPAD  64                       // NPTS*3=39 padded to a 64-col GEMM

typedef __attribute__((ext_vector_type(16))) __bf16 v16bf;
typedef __attribute__((ext_vector_type(8)))  float  v8f;

union FragU { uint4 q[2]; v16bf v; };

__device__ __forceinline__ unsigned short f2bf(float f) {
  union { float f; unsigned u; } x; x.f = f;
  unsigned u = x.u;
  unsigned r = u + 0x7FFFu + ((u >> 16) & 1u);   // round-to-nearest-even
  return (unsigned short)(r >> 16);
}
__device__ __forceinline__ float bf2f(unsigned short h) {
  union { unsigned u; float f; } x; x.u = ((unsigned)h) << 16;
  return x.f;
}

// ---------------------------------------------------------------------------
// bf16 conversion of instance_feature and (instance_feature + anchor_embed)
// ---------------------------------------------------------------------------
__global__ void prep_feat(const float* __restrict__ inst, const float* __restrict__ emb,
                          unsigned short* __restrict__ instB, unsigned short* __restrict__ featB) {
  int t = blockIdx.x * blockDim.x + threadIdx.x;
  if (t < M_ROWS * E_DIM) {
    float a = inst[t];
    instB[t] = f2bf(a);
    featB[t] = f2bf(a + emb[t]);
  }
}

// ---------------------------------------------------------------------------
// Weight prep: W (K x N, f32) -> WT (Npad x K, bf16), zero-padded cols;
// optional zero-padded bias copy.
// ---------------------------------------------------------------------------
__global__ void prep_wT(const float* __restrict__ W, const float* __restrict__ bias,
                        unsigned short* __restrict__ WT, float* __restrict__ biasPad,
                        int K, int N, int Npad) {
  int t = blockIdx.x * blockDim.x + threadIdx.x;
  if (t < Npad * K) {
    int k = t % K, n = t / K;
    float v = (n < N) ? W[(size_t)k * N + n] : 0.f;
    WT[t] = f2bf(v);
  }
  if (biasPad && t < Npad) biasPad[t] = (t < N) ? bias[t] : 0.f;
}

// ---------------------------------------------------------------------------
// NCHW f32 -> NHWC bf16 transpose of a feature level (LDS-tiled, coalesced
// both directions). grid = (ceil(HW/64), 4, 12), block = 256.
// ---------------------------------------------------------------------------
__global__ __launch_bounds__(256)
void transpose_fm_bf16(const float* __restrict__ in, unsigned short* __restrict__ out, int HW) {
  __shared__ float tile[64][65];
  const int img  = blockIdx.z;
  const int pos0 = blockIdx.x * 64, c0 = blockIdx.y * 64;
  const int tx = threadIdx.x & 63, ty = threadIdx.x >> 6; // ty in 0..3
#pragma unroll
  for (int i = 0; i < 16; ++i) {
    int c = c0 + ty + i * 4;
    int pos = pos0 + tx;
    float v = (pos < HW) ? in[((size_t)img * E_DIM + c) * HW + pos] : 0.f;
    tile[ty + i * 4][tx] = v;
  }
  __syncthreads();
#pragma unroll
  for (int i = 0; i < 16; ++i) {
    int pos = pos0 + ty + i * 4;
    int c = c0 + tx;
    if (pos < HW) out[((size_t)img * HW + pos) * E_DIM + c] = f2bf(tile[tx][ty + i * 4]);
  }
}

// ---------------------------------------------------------------------------
// bf16 WMMA GEMM: D(MxN,f32) = A(MxK,bf16) * BT(NxK,bf16)^T + bias[n]
//                              (+ residual[m,n]).
// REQUIRES: N % 64 == 0, K % 32 == 0 (caller pads). One wave computes a
// 32x64 tile: 2 A-fragments x 4 B-fragments = 8 unconditional
// v_wmma_f32_16x16x32_bf16 per K step (branch-free inner loop -> accumulators
// stay pinned, no phi-copy storms). Block = 128 threads = 4 waves = 128 rows.
// M edge: per-lane clamped loads + guarded stores; the early wave exit is
// wave-uniform so EXEC is all-ones at every WMMA.
// Fragment layout per CDNA5 16-bit A-matrix spec: per lane (r = lane&15,
// h = lane>>4), elements 0..7 = K[h*8 .. h*8+7], 8..15 = K[16+h*8 ..
// 16+h*8+7]; two contiguous 16B loads per operand per K step (BT is N x K).
// ---------------------------------------------------------------------------
__global__ __launch_bounds__(128)
void gemm_wmma_bf16(const unsigned short* __restrict__ A,
                    const unsigned short* __restrict__ BT,
                    const float* __restrict__ bias,
                    const float* __restrict__ residual,
                    float* __restrict__ D,
                    int M, int N, int K) {
  const int lane = threadIdx.x & 31;
  const int wave = threadIdx.x >> 5;
  const int m0 = (blockIdx.y * 4 + wave) << 5;   // 32 rows per wave
  if (m0 >= M) return;                           // wave-uniform exit
  const int n0  = blockIdx.x << 6;               // 64 cols per wave
  const int h   = lane >> 4;
  const int l16 = lane & 15;

  int mr0 = m0 + l16;       if (mr0 > M - 1) mr0 = M - 1;
  int mr1 = m0 + 16 + l16;  if (mr1 > M - 1) mr1 = M - 1;
  const unsigned short* Arow0 = A + (size_t)mr0 * K;
  const unsigned short* Arow1 = A + (size_t)mr1 * K;
  const unsigned short* Brow[4];
#pragma unroll
  for (int t = 0; t < 4; ++t)
    Brow[t] = BT + (size_t)(n0 + (t << 4) + l16) * K;

  v8f acc[2][4] = {};
  for (int kb = 0; kb < K; kb += 32) {
    FragU a0, a1;
    a0.q[0] = *reinterpret_cast<const uint4*>(Arow0 + kb + h * 8);
    a0.q[1] = *reinterpret_cast<const uint4*>(Arow0 + kb + 16 + h * 8);
    a1.q[0] = *reinterpret_cast<const uint4*>(Arow1 + kb + h * 8);
    a1.q[1] = *reinterpret_cast<const uint4*>(Arow1 + kb + 16 + h * 8);
#pragma unroll
    for (int t = 0; t < 4; ++t) {
      FragU bfrag;
      bfrag.q[0] = *reinterpret_cast<const uint4*>(Brow[t] + kb + h * 8);
      bfrag.q[1] = *reinterpret_cast<const uint4*>(Brow[t] + kb + 16 + h * 8);
      acc[0][t] = __builtin_amdgcn_wmma_f32_16x16x32_bf16(
          false, a0.v, false, bfrag.v, (short)0, acc[0][t], false, false);
      acc[1][t] = __builtin_amdgcn_wmma_f32_16x16x32_bf16(
          false, a1.v, false, bfrag.v, (short)0, acc[1][t], false, false);
    }
  }

  // C/D layout: VGPR v -> row subtile_m0 + v + 8*h, col n0 + t*16 + (lane&15)
#pragma unroll
  for (int s = 0; s < 2; ++s) {
#pragma unroll
    for (int t = 0; t < 4; ++t) {
      int n = n0 + (t << 4) + l16;               // always < N (N % 64 == 0)
      float bn = bias ? bias[n] : 0.f;
#pragma unroll
      for (int v = 0; v < 8; ++v) {
        int m = m0 + s * 16 + v + 8 * h;
        if (m < M) {
          size_t idx = (size_t)m * N + n;
          float val = acc[s][t][v] + bn;
          if (residual) val += residual[idx];
          D[idx] = val;
        }
      }
    }
  }
}

// ---------------------------------------------------------------------------
// Project key points to normalized grid coords. One thread per (b,c,n,p).
// ---------------------------------------------------------------------------
__global__ void project_kernel(const float* __restrict__ anchor, const float* __restrict__ offs,
                               const float* __restrict__ proj, float* __restrict__ gridArr) {
  int t = blockIdx.x * blockDim.x + threadIdx.x;
  if (t >= BS * NCAMS * NA_Q * NPTS) return;
  int p = t % NPTS; int r = t / NPTS;
  int n = r % NA_Q; r /= NA_Q;
  int c = r % NCAMS; int b = r / NCAMS;
  const float* av = anchor + ((size_t)b * NA_Q + n) * E_DIM;
  const float* ov = offs + ((size_t)b * NA_Q + n) * KPAD + p * 3;
  float kx = av[0] + ov[0], ky = av[1] + ov[1], kz = av[2] + ov[2];
  const float* Pm = proj + (size_t)(b * NCAMS + c) * 16;
  float px = Pm[0] * kx + Pm[1] * ky + Pm[2]  * kz + Pm[3];
  float py = Pm[4] * kx + Pm[5] * ky + Pm[6]  * kz + Pm[7];
  float pz = Pm[8] * kx + Pm[9] * ky + Pm[10] * kz + Pm[11];
  float inv = 1.0f / fmaxf(pz, 1e-5f);
  gridArr[(size_t)t * 2 + 0] = px * inv;
  gridArr[(size_t)t * 2 + 1] = py * inv;
}

// ---------------------------------------------------------------------------
// In-place softmax over the 312 slots per (row, group). One wave per block.
// ---------------------------------------------------------------------------
__global__ void softmax_slots(float* __restrict__ logits) {
  int row = blockIdx.x >> 3;
  int g   = blockIdx.x & 7;
  int lane = threadIdx.x;
  float* base = logits + (size_t)row * NWFC + g;
  float mx = -3.4e38f;
  for (int j = lane; j < NSLOT; j += 32) mx = fmaxf(mx, base[(size_t)j * NGRP]);
#pragma unroll
  for (int o = 16; o > 0; o >>= 1) mx = fmaxf(mx, __shfl_xor(mx, o, 32));
  float sum = 0.f;
  for (int j = lane; j < NSLOT; j += 32) {
    float ev = expf(base[(size_t)j * NGRP] - mx);
    base[(size_t)j * NGRP] = ev;
    sum += ev;
  }
#pragma unroll
  for (int o = 16; o > 0; o >>= 1) sum += __shfl_xor(sum, o, 32);
  float inv = 1.f / sum;
  for (int j = lane; j < NSLOT; j += 32) base[(size_t)j * NGRP] *= inv;
}

// ---------------------------------------------------------------------------
// Fused bilinear sampling + grouped weighted reduction. One block per (b,n),
// one thread per channel e. Slot params + softmax weights staged in LDS.
// NHWC-bf16 path: each corner read is 512B contiguous across the block.
// ---------------------------------------------------------------------------
__global__ __launch_bounds__(256)
void sample_fuse(const float* __restrict__ gridArr, const float* __restrict__ wsm,
                 const unsigned short* __restrict__ t0, const unsigned short* __restrict__ t1,
                 const unsigned short* __restrict__ t2, const unsigned short* __restrict__ t3,
                 const float* __restrict__ f0, const float* __restrict__ f1,
                 const float* __restrict__ f2, const float* __restrict__ f3,
                 int useT, unsigned short* __restrict__ fusedB) {
  __shared__ float s_x0[NSLOT], s_y0[NSLOT], s_fx[NSLOT], s_fy[NSLOT];
  __shared__ float s_w[NWFC];
  const int bn = blockIdx.x;
  const int b = bn / NA_Q, n = bn % NA_Q;
  const int e = threadIdx.x;
  const int Wl[4] = {176, 88, 44, 22};
  const int Hl[4] = {64, 32, 16, 8};

  for (int i = e; i < NWFC; i += 256) s_w[i] = wsm[(size_t)bn * NWFC + i];
  for (int s = e; s < NSLOT; s += 256) {
    int c = s / (NLVLS * NPTS), r = s % (NLVLS * NPTS), l = r / NPTS, p = r % NPTS;
    const float* g = gridArr + (size_t)(((b * NCAMS + c) * NA_Q + n) * NPTS + p) * 2;
    float ix = ((g[0] + 1.f) * Wl[l] - 1.f) * 0.5f;
    float iy = ((g[1] + 1.f) * Hl[l] - 1.f) * 0.5f;
    float fx = floorf(ix), fy = floorf(iy);
    s_x0[s] = fx; s_y0[s] = fy; s_fx[s] = ix - fx; s_fy[s] = iy - fy;
  }
  __syncthreads();

  const int g8 = e >> 5;
  const unsigned short* tl[4] = {t0, t1, t2, t3};
  const float* fl[4] = {f0, f1, f2, f3};
  float acc = 0.f;
  int slot = 0;
  for (int c = 0; c < NCAMS; ++c) {
    const int img = b * NCAMS + c;
    for (int l = 0; l < NLVLS; ++l) {
      const int W = Wl[l], H = Hl[l], HW = W * H;
      const unsigned short* tb = useT ? (tl[l] + (size_t)img * HW * E_DIM + e) : (const unsigned short*)0;
      const float* fb = fl[l] + (size_t)(img * E_DIM + e) * HW;
      for (int p = 0; p < NPTS; ++p, ++slot) {
        float wv = s_w[slot * NGRP + g8];
        float fx = s_fx[slot], fy = s_fy[slot];
        int x0 = (int)s_x0[slot], y0 = (int)s_y0[slot];
        bool vx0 = (unsigned)x0 < (unsigned)W;
        bool vx1 = (unsigned)(x0 + 1) < (unsigned)W;
        bool vy0 = (unsigned)y0 < (unsigned)H;
        bool vy1 = (unsigned)(y0 + 1) < (unsigned)H;
        int x0c = vx0 ? x0 : 0, x1c = vx1 ? x0 + 1 : 0;
        int y0c = vy0 ? y0 : 0, y1c = vy1 ? y0 + 1 : 0;
        float v00, v01, v10, v11;
        if (useT) {
          v00 = (vy0 && vx0) ? bf2f(tb[(size_t)(y0c * W + x0c) * E_DIM]) : 0.f;
          v01 = (vy0 && vx1) ? bf2f(tb[(size_t)(y0c * W + x1c) * E_DIM]) : 0.f;
          v10 = (vy1 && vx0) ? bf2f(tb[(size_t)(y1c * W + x0c) * E_DIM]) : 0.f;
          v11 = (vy1 && vx1) ? bf2f(tb[(size_t)(y1c * W + x1c) * E_DIM]) : 0.f;
        } else {
          v00 = (vy0 && vx0) ? fb[y0c * W + x0c] : 0.f;
          v01 = (vy0 && vx1) ? fb[y0c * W + x1c] : 0.f;
          v10 = (vy1 && vx0) ? fb[y1c * W + x0c] : 0.f;
          v11 = (vy1 && vx1) ? fb[y1c * W + x1c] : 0.f;
        }
        acc += wv * ((1.f - fy) * ((1.f - fx) * v00 + fx * v01)
                   +        fy  * ((1.f - fx) * v10 + fx * v11));
      }
    }
  }
  fusedB[(size_t)bn * E_DIM + e] = f2bf(acc);
}

// ---------------------------------------------------------------------------
// host-side orchestration
// ---------------------------------------------------------------------------
extern "C" void kernel_launch(void* const* d_in, const int* in_sizes, int n_in,
                              void* d_out, int out_size, void* d_ws, size_t ws_size,
                              hipStream_t stream) {
  const float* inst   = (const float*)d_in[0];
  const float* anchor = (const float*)d_in[1];
  const float* emb    = (const float*)d_in[2];
  const float* fm[4]  = {(const float*)d_in[3], (const float*)d_in[4],
                         (const float*)d_in[5], (const float*)d_in[6]};
  const float* proj   = (const float*)d_in[7];
  const float* Wkps   = (const float*)d_in[8];
  const float* bkps   = (const float*)d_in[9];
  const float* Wwfc   = (const float*)d_in[10];
  const float* bwfc   = (const float*)d_in[11];
  const float* Wout   = (const float*)d_in[12];
  const float* bout   = (const float*)d_in[13];
  float* out = (float*)d_out;

  const int HWs[4] = {64 * 176, 32 * 88, 16 * 44, 8 * 22};

  // workspace layout (256B aligned)
  size_t off = 0;
  auto alloc = [&](size_t bytes) -> size_t {
    size_t o = off; off += (bytes + 255) & ~(size_t)255; return o;
  };
  char* wsb = (char*)d_ws;
  unsigned short* instB = (unsigned short*)(wsb + alloc((size_t)M_ROWS * E_DIM * 2));
  unsigned short* featB = (unsigned short*)(wsb + alloc((size_t)M_ROWS * E_DIM * 2));
  unsigned short* WkpsT = (unsigned short*)(wsb + alloc((size_t)KPAD * E_DIM * 2));
  unsigned short* WwfcT = (unsigned short*)(wsb + alloc((size_t)NWFC * E_DIM * 2));
  unsigned short* WoutT = (unsigned short*)(wsb + alloc((size_t)E_DIM * E_DIM * 2));
  float* biasKps = (float*)(wsb + alloc((size_t)KPAD * 4));
  float* offs    = (float*)(wsb + alloc((size_t)M_ROWS * KPAD * 4));
  float* logits  = (float*)(wsb + alloc((size_t)M_ROWS * NWFC * 4));
  float* gridArr = (float*)(wsb + alloc((size_t)BS * NCAMS * NA_Q * NPTS * 2 * 4));
  unsigned short* fusedB = (unsigned short*)(wsb + alloc((size_t)M_ROWS * E_DIM * 2));
  unsigned short* fmT[4];
  for (int l = 0; l < 4; ++l)
    fmT[l] = (unsigned short*)(wsb + alloc((size_t)12 * HWs[l] * E_DIM * 2));
  const int useT = (off <= ws_size) ? 1 : 0;   // NHWC-bf16 fast path if ws fits (~116 MB)

  // 1) bf16 activations
  prep_feat<<<(M_ROWS * E_DIM + 255) / 256, 256, 0, stream>>>(inst, emb, instB, featB);
  // 2) weights -> transposed bf16 (+ padded kps bias)
  prep_wT<<<(KPAD * E_DIM + 255) / 256, 256, 0, stream>>>(Wkps, bkps, WkpsT, biasKps, E_DIM, NPTS * 3, KPAD);
  prep_wT<<<(NWFC * E_DIM + 255) / 256, 256, 0, stream>>>(Wwfc, nullptr, WwfcT, nullptr, E_DIM, NWFC, NWFC);
  prep_wT<<<(E_DIM * E_DIM + 255) / 256, 256, 0, stream>>>(Wout, nullptr, WoutT, nullptr, E_DIM, E_DIM, E_DIM);
  // 3) feature maps NCHW f32 -> NHWC bf16 (L2-resident: 92 MB < 192 MB L2)
  if (useT) {
    for (int l = 0; l < 4; ++l)
      transpose_fm_bf16<<<dim3((HWs[l] + 63) / 64, 4, 12), 256, 0, stream>>>(fm[l], fmT[l], HWs[l]);
  }
  const dim3 gBlk(128);
  const int gy = (M_ROWS + 127) / 128;   // 4 waves/block, 32 rows/wave
  // 4) kps GEMM -> offs (1800 x 64, first 39 cols used)
  gemm_wmma_bf16<<<dim3(KPAD / 64, gy), gBlk, 0, stream>>>(
      instB, WkpsT, biasKps, nullptr, offs, M_ROWS, KPAD, E_DIM);
  // 5) projection -> grid coords
  project_kernel<<<(BS * NCAMS * NA_Q * NPTS + 255) / 256, 256, 0, stream>>>(
      anchor, offs, proj, gridArr);
  // 6) wfc GEMM -> logits (1800 x 2496), then per-group softmax in place
  gemm_wmma_bf16<<<dim3(NWFC / 64, gy), gBlk, 0, stream>>>(
      featB, WwfcT, bwfc, nullptr, logits, M_ROWS, NWFC, E_DIM);
  softmax_slots<<<M_ROWS * NGRP, 32, 0, stream>>>(logits);
  // 7) fused bilinear sampling + grouped weighted reduction -> fused (bf16)
  sample_fuse<<<M_ROWS, 256, 0, stream>>>(
      gridArr, logits,
      useT ? fmT[0] : nullptr, useT ? fmT[1] : nullptr,
      useT ? fmT[2] : nullptr, useT ? fmT[3] : nullptr,
      fm[0], fm[1], fm[2], fm[3], useT, fusedB);
  // 8) output GEMM + bias + residual -> d_out
  gemm_wmma_bf16<<<dim3(E_DIM / 64, gy), gBlk, 0, stream>>>(
      fusedB, WoutT, bout, inst, out, M_ROWS, E_DIM, E_DIM);
}
